// AttentionHead_18734647345222
// MI455X (gfx1250) — matrix operand order repaired and checked by reading, not compiled, not run on hardware
//
#include <hip/hip_runtime.h>
#include <hip/hip_bf16.h>

typedef __bf16 bf16_t;
typedef __attribute__((ext_vector_type(16))) __bf16 v16bf;
typedef __attribute__((ext_vector_type(8)))  __bf16 v8bf;
typedef __attribute__((ext_vector_type(4)))  __bf16 v4bf;
typedef __attribute__((ext_vector_type(8)))  float  v8f;
typedef __attribute__((ext_vector_type(4)))  unsigned int v4u;
typedef __attribute__((ext_vector_type(8)))  int v8i;
typedef __attribute__((ext_vector_type(4)))  int v4i;

#define BATCH   4
#define SEQ     2048
#define D_MODEL 1024
#define D_HEAD  64
#define QTILE   32      // queries per block in the attention kernel
#define KBLK    32      // keys per iteration

// ---------------------------------------------------------------------------
// WMMA fragment helpers (CDNA5 16x16x32 bf16, wave32)
// A (16x32, 16-bit): lane holds row M=lane&15; e<8 -> K=e+8h, e>=8 -> K=e+8+8h
//   (h = lane>>4)  => two contiguous 16B loads at k0+8h and k0+8h+16.
// B (32x16, 16-bit): lane holds col N=lane&15; same K order as A
//   => two contiguous 16B loads at k0+8h and k0+8h+16.
// C/D: acc[r] is element (M = r + 8h, N = lane&15).
// ---------------------------------------------------------------------------
__device__ __forceinline__ v16bf load_a_frag(const bf16_t* __restrict__ base,
                                             int ld, int row, int k0, int lane) {
  const int h = (lane >> 4) & 1;
  const bf16_t* p = base + (size_t)row * ld + k0 + 8 * h;
  v8bf lo = *(const v8bf*)(p);
  v8bf hi = *(const v8bf*)(p + 16);
  v16bf r;
#pragma unroll
  for (int e = 0; e < 8; ++e) { r[e] = lo[e]; r[e + 8] = hi[e]; }
  return r;
}

__device__ __forceinline__ v16bf load_b_frag(const bf16_t* __restrict__ base,
                                             int ld, int col, int k0, int lane) {
  const int h = (lane >> 4) & 1;
  const bf16_t* p = base + (size_t)col * ld + k0 + 8 * h;
  v8bf lo = *(const v8bf*)(p);
  v8bf hi = *(const v8bf*)(p + 16);
  v16bf r;
#pragma unroll
  for (int e = 0; e < 8; ++e) { r[e] = lo[e]; r[e + 8] = hi[e]; }
  return r;
}

__device__ __forceinline__ v8f wmma_bf16(v16bf a, v16bf b, v8f c) {
  return __builtin_amdgcn_wmma_f32_16x16x32_bf16(
      false, a, false, b, (short)0, c, false, false);
}

// ---------------------------------------------------------------------------
// TDM: async DMA of one V block (2-D tile: 32 keys x 1024 d-rows of bf16,
// row ld = 2048 keys) from global into LDS (row-major ld = 32).
// D# packing per CDNA5 ISA 8.3-8.5: group0 = {count=1, lds_addr,
// global_addr[56:0], type=2}; group1 = {data_size=2B, tensor_dim0=2048,
// tensor_dim1=1024, tile_dim0=32, tile_dim1=1024, dim0_stride=2048};
// remaining groups zero (tensor is 2-D). Completion tracked with TENSORcnt.
// This toolchain exposes the 6-arg builtin:
//   (uint32x4, int32x8, int32x4, int32x4, int32x8, i32 cpol)
// ---------------------------------------------------------------------------
__device__ __forceinline__ void tdm_load_v_block(const bf16_t* gsrc,
                                                 unsigned lds_byte_off) {
  const unsigned long long ga = (unsigned long long)(uintptr_t)gsrc;
  const unsigned td0 = SEQ;          // tensor dim0 length (keys)
  const unsigned td1 = D_MODEL;      // tensor dim1 length (d rows)
  const unsigned tl0 = KBLK;         // tile dim0 (keys)
  const unsigned tl1 = D_MODEL;      // tile dim1 (d rows)
  const unsigned s0  = SEQ;          // dim0 stride (elements)
  v4u g0 = { 1u,                                            // count=1
             lds_byte_off,                                  // lds_addr
             (unsigned)ga,                                  // global_addr[31:0]
             (unsigned)((ga >> 32) & 0x01FFFFFFu) | (2u << 30) };  // [56:32]|type=2
  v8i g1 = { (int)0x00010000u,                              // data_size=1 (2B)
             (int)((td0 & 0xFFFFu) << 16),                  // tensor_dim0[15:0]
             (int)(((td0 >> 16) & 0xFFFFu) | ((td1 & 0xFFFFu) << 16)),
             (int)(((td1 >> 16) & 0xFFFFu) | ((tl0 & 0xFFFFu) << 16)),
             (int)(tl1 & 0xFFFFu),                          // tile_dim1 (tile_dim2=0)
             (int)s0,                                       // dim0_stride[31:0]
             0, 0 };
  v4i g2 = { 0, 0, 0, 0 };
  v4i g3 = { 0, 0, 0, 0 };
  v8i g4 = { 0, 0, 0, 0, 0, 0, 0, 0 };
  __builtin_amdgcn_tensor_load_to_lds(g0, g1, g2, g3, g4, 0);
}

// ---------------------------------------------------------------------------
// Pass 0a: f32 -> bf16 (4-wide)
// ---------------------------------------------------------------------------
__global__ void k_cvt_bf16(const float* __restrict__ in, bf16_t* __restrict__ out,
                           int n4) {
  for (int i = blockIdx.x * blockDim.x + threadIdx.x; i < n4;
       i += gridDim.x * blockDim.x) {
    float4 v = ((const float4*)in)[i];
    v4bf o;
    o[0] = (bf16_t)v.x; o[1] = (bf16_t)v.y; o[2] = (bf16_t)v.z; o[3] = (bf16_t)v.w;
    ((v4bf*)out)[i] = o;
  }
}

// ---------------------------------------------------------------------------
// Pass 0b: transpose + convert  Wt[n][k] = bf16(W[k][n]),  W: [K][N]
// ---------------------------------------------------------------------------
__global__ void k_transpose_cvt(const float* __restrict__ W, bf16_t* __restrict__ Wt,
                                int K, int N) {
  int total = K * N;
  for (int i = blockIdx.x * blockDim.x + threadIdx.x; i < total;
       i += gridDim.x * blockDim.x) {
    int n = i / K;
    int k = i - n * K;
    Wt[i] = (bf16_t)W[(size_t)k * N + n];
  }
}

// ---------------------------------------------------------------------------
// Pass 1: projection GEMM, 16x64 tile per wave (A-fragment reused over 4
// B-tiles: 10 b128 loads per 4 WMMAs). blockIdx: x = 64-col group,
// y = 16-row tile, z = batch.
// transpose_out=0 : out[b][m][n] (ld=Ncols)   (Q, K)
// transpose_out=1 : out[b][n][m] (ld=SEQ)     (V transposed)
// ---------------------------------------------------------------------------
__global__ __launch_bounds__(32) void k_proj(
    const bf16_t* __restrict__ xh, const bf16_t* __restrict__ wt,
    const float* __restrict__ bias, bf16_t* __restrict__ out,
    int Ncols, int transpose_out) {
  const int lane = threadIdx.x & 31;
  const int ct = blockIdx.x, rt = blockIdx.y, b = blockIdx.z;
  const bf16_t* xb = xh + (size_t)b * SEQ * D_MODEL;

  const int nlo = lane & 15;
  const int h = (lane >> 4) & 1;
  const int arow = rt * 16 + nlo;

  v8f acc[4];
#pragma unroll
  for (int t = 0; t < 4; ++t) acc[t] = (v8f){};

#pragma unroll 2
  for (int k = 0; k < D_MODEL; k += 32) {
    v16bf a = load_a_frag(xb, D_MODEL, arow, k, lane);
#pragma unroll
    for (int t = 0; t < 4; ++t) {
      v16bf bb = load_b_frag(wt, D_MODEL, ct * 64 + t * 16 + nlo, k, lane);
      acc[t] = wmma_bf16(a, bb, acc[t]);
    }
  }

#pragma unroll
  for (int t = 0; t < 4; ++t) {
    const int n = ct * 64 + t * 16 + nlo;
    const float bv = bias[n];
#pragma unroll
    for (int r = 0; r < 8; ++r) {
      const int m = rt * 16 + r + 8 * h;
      const float val = acc[t][r] + bv;
      if (!transpose_out)
        out[(size_t)b * SEQ * Ncols + (size_t)m * Ncols + n] = (bf16_t)val;
      else
        out[(size_t)b * Ncols * SEQ + (size_t)n * SEQ + m] = (bf16_t)val;
    }
  }
}

// ---------------------------------------------------------------------------
// Pass 2: fused causal flash attention, 32-query tiles.
// Block = 256 threads (8 waves) per (batch, 32-query tile).
//   wave 0: issues TDM for the NEXT V block (64KB -> double-buffered LDS),
//           waits TENSORcnt, and does softmax for query subtile 0.
//   wave 1: softmax for query subtile 1 (concurrently).
//   all waves: rescale O, O += P @ V (16 WMMAs over a 256-col D slice),
//           V fragments come from LDS (ds_load_b128).
// ---------------------------------------------------------------------------
__global__ __launch_bounds__(256, 1) void k_attn(
    const bf16_t* __restrict__ qh, const bf16_t* __restrict__ kh,
    const bf16_t* __restrict__ vT, float* __restrict__ out) {
  const int tid  = threadIdx.x;
  const int lane = tid & 31;
  const int wave = tid >> 5;
  const int b    = blockIdx.y;
  const int qbase = blockIdx.x * QTILE;
  const int h   = (lane >> 4) & 1;
  const int nlo = lane & 15;
  const int qs    = wave & 1;          // query subtile this wave accumulates
  const int dbase = (wave >> 1) * 256; // 256-column D slice (16 tiles)

  const bf16_t* qb = qh + (size_t)b * SEQ * D_HEAD;
  const bf16_t* kb = kh + (size_t)b * SEQ * D_HEAD;
  const bf16_t* vb = vT + (size_t)b * D_MODEL * SEQ;

  extern __shared__ char smem[];                       // 2 x 64KB V buffers
  bf16_t* vls = (bf16_t*)smem;                         // [buf][d*KBLK + key]
  __shared__ __align__(16) bf16_t Pls[2][16 * KBLK];   // P tiles, ld=KBLK
  __shared__ float alpha_ls[2][16];
  __shared__ float l_ls[2][16];

  v8f O[16];
#pragma unroll
  for (int t = 0; t < 16; ++t) O[t] = (v8f){};

  // softmax state for waves 0/1 (per-lane rows: qrow0 + r + 8h)
  float m_run[8], l_run[8];
#pragma unroll
  for (int r = 0; r < 8; ++r) { m_run[r] = -__builtin_inff(); l_run[r] = 0.f; }

  const int qrow0 = qbase + qs * 16;   // for waves 0/1: their subtile row base
  v16bf qa0, qa1;
  if (wave < 2) {
    qa0 = load_a_frag(qb, D_HEAD, qrow0 + nlo, 0, lane);
    qa1 = load_a_frag(qb, D_HEAD, qrow0 + nlo, 32, lane);
  }

  const int nb = (qbase + QTILE + 31) >> 5;            // causal key blocks

  if (wave == 0)  // prefetch first V block via TDM
    tdm_load_v_block(vb + 0, (unsigned)(uintptr_t)(vls));

  for (int j = 0; j < nb; ++j) {
    if (wave == 0) {
      // pipeline: kick off next block's DMA, then wait for current block
      if (j + 1 < nb) {
        tdm_load_v_block(vb + (size_t)(j + 1) * KBLK,
                         (unsigned)(uintptr_t)(vls + ((j + 1) & 1) * D_MODEL * KBLK));
        __builtin_amdgcn_s_wait_tensorcnt(1);
      } else {
        __builtin_amdgcn_s_wait_tensorcnt(0);
      }
    }
    if (wave < 2) {
      // ---- scores S = Q.K^T for this wave's 16-query subtile
      v8f S0 = {}, S1 = {};
      {
        v16bf b0 = load_b_frag(kb, D_HEAD, j * KBLK + nlo, 0, lane);
        v16bf b1 = load_b_frag(kb, D_HEAD, j * KBLK + nlo, 32, lane);
        S0 = wmma_bf16(qa0, b0, S0);
        S0 = wmma_bf16(qa1, b1, S0);
      }
      {
        v16bf b0 = load_b_frag(kb, D_HEAD, j * KBLK + 16 + nlo, 0, lane);
        v16bf b1 = load_b_frag(kb, D_HEAD, j * KBLK + 16 + nlo, 32, lane);
        S1 = wmma_bf16(qa0, b0, S1);
        S1 = wmma_bf16(qa1, b1, S1);
      }
      if (j + 1 < nb)
        __builtin_prefetch(kb + (size_t)((j + 1) * KBLK + nlo) * D_HEAD, 0, 1);

      // ---- causal mask on diagonal-touching blocks
      if (j * KBLK + KBLK - 1 > qrow0) {
        const int n0 = j * KBLK + nlo, n1 = n0 + 16;
#pragma unroll
        for (int r = 0; r < 8; ++r) {
          const int m = qrow0 + r + 8 * h;
          if (n0 > m) S0[r] = -__builtin_inff();
          if (n1 > m) S1[r] = -__builtin_inff();
        }
      }
      // ---- row max (xor <=8 stays within each 16-lane half)
      float t8[8];
#pragma unroll
      for (int r = 0; r < 8; ++r) t8[r] = fmaxf(S0[r], S1[r]);
#pragma unroll
      for (int msk = 8; msk >= 1; msk >>= 1)
#pragma unroll
        for (int r = 0; r < 8; ++r) t8[r] = fmaxf(t8[r], __shfl_xor(t8[r], msk, 32));

      float P0[8], P1[8], rs[8];
#pragma unroll
      for (int r = 0; r < 8; ++r) {
        const float mn = fmaxf(m_run[r], t8[r]);
        const float al = __expf(m_run[r] - mn);
        m_run[r] = mn;
        t8[r] = al;                      // alpha
        P0[r] = __expf(S0[r] - mn);
        P1[r] = __expf(S1[r] - mn);
        rs[r] = P0[r] + P1[r];
      }
#pragma unroll
      for (int msk = 8; msk >= 1; msk >>= 1)
#pragma unroll
        for (int r = 0; r < 8; ++r) rs[r] += __shfl_xor(rs[r], msk, 32);
#pragma unroll
      for (int r = 0; r < 8; ++r) l_run[r] = l_run[r] * t8[r] + rs[r];

      if (nlo == 0) {
#pragma unroll
        for (int r = 0; r < 8; ++r) alpha_ls[qs][r + 8 * h] = t8[r];
      }
#pragma unroll
      for (int r = 0; r < 8; ++r) {
        Pls[qs][(r + 8 * h) * KBLK + nlo]      = (bf16_t)P0[r];
        Pls[qs][(r + 8 * h) * KBLK + nlo + 16] = (bf16_t)P1[r];
      }
    }
    __syncthreads();

    // ---- all waves: rescale O, then O += P @ V from LDS
    float a8[8];
#pragma unroll
    for (int r = 0; r < 8; ++r) a8[r] = alpha_ls[qs][r + 8 * h];
#pragma unroll
    for (int t = 0; t < 16; ++t)
#pragma unroll
      for (int r = 0; r < 8; ++r) O[t][r] *= a8[r];

    v16bf pa = load_a_frag(Pls[qs], KBLK, nlo, 0, lane);
    const bf16_t* vcur = vls + (j & 1) * D_MODEL * KBLK;
#pragma unroll
    for (int t = 0; t < 16; ++t) {
      v16bf vbf = load_b_frag(vcur, KBLK, dbase + t * 16 + nlo, 0, lane);
      O[t] = wmma_bf16(pa, vbf, O[t]);
    }
    __syncthreads();  // protect P/alpha (and buffer rotation) for next iter
  }

  // ---- epilogue: normalize and store f32
  if (wave < 2 && nlo == 0) {
#pragma unroll
    for (int r = 0; r < 8; ++r) l_ls[qs][r + 8 * h] = l_run[r];
  }
  __syncthreads();

  float linv[8];
#pragma unroll
  for (int r = 0; r < 8; ++r) linv[r] = 1.0f / l_ls[qs][r + 8 * h];

  float* ob = out + (size_t)b * SEQ * D_MODEL;
#pragma unroll
  for (int t = 0; t < 16; ++t) {
    const int d = dbase + t * 16 + nlo;
#pragma unroll
    for (int r = 0; r < 8; ++r) {
      const int m = qbase + qs * 16 + r + 8 * h;
      ob[(size_t)m * D_MODEL + d] = O[t][r] * linv[r];
    }
  }
}

// ---------------------------------------------------------------------------
// Launcher
// ---------------------------------------------------------------------------
extern "C" void kernel_launch(void* const* d_in, const int* in_sizes, int n_in,
                              void* d_out, int out_size, void* d_ws, size_t ws_size,
                              hipStream_t stream) {
  const float* x   = (const float*)d_in[0];
  const float* WQ  = (const float*)d_in[1];
  const float* bQ  = (const float*)d_in[2];
  const float* WK  = (const float*)d_in[3];
  const float* bK  = (const float*)d_in[4];
  const float* WOV = (const float*)d_in[5];
  const float* bOV = (const float*)d_in[6];
  float* out = (float*)d_out;

  char* ws = (char*)d_ws;
  size_t off = 0;
  bf16_t* xh   = (bf16_t*)(ws + off); off += (size_t)BATCH * SEQ * D_MODEL * 2;   // 16 MB
  bf16_t* wqt  = (bf16_t*)(ws + off); off += (size_t)D_HEAD * D_MODEL * 2;        // 128 KB
  bf16_t* wkt  = (bf16_t*)(ws + off); off += (size_t)D_HEAD * D_MODEL * 2;        // 128 KB
  bf16_t* wovt = (bf16_t*)(ws + off); off += (size_t)D_MODEL * D_MODEL * 2;       // 2 MB
  bf16_t* qhp  = (bf16_t*)(ws + off); off += (size_t)BATCH * SEQ * D_HEAD * 2;    // 1 MB
  bf16_t* khp  = (bf16_t*)(ws + off); off += (size_t)BATCH * SEQ * D_HEAD * 2;    // 1 MB
  bf16_t* vtp  = (bf16_t*)(ws + off); off += (size_t)BATCH * D_MODEL * SEQ * 2;   // 16 MB

  // Pass 0: precision conversion + weight transposes
  k_cvt_bf16<<<2048, 256, 0, stream>>>(x, xh, BATCH * SEQ * D_MODEL / 4);
  k_transpose_cvt<<<256, 256, 0, stream>>>(WQ, wqt, D_MODEL, D_HEAD);
  k_transpose_cvt<<<256, 256, 0, stream>>>(WK, wkt, D_MODEL, D_HEAD);
  k_transpose_cvt<<<2048, 256, 0, stream>>>(WOV, wovt, D_MODEL, D_MODEL);

  // Pass 1: projections (WMMA GEMMs, 16x64 per wave)
  k_proj<<<dim3(D_HEAD / 64, SEQ / 16, BATCH), 32, 0, stream>>>(xh, wqt, bQ, qhp, D_HEAD, 0);
  k_proj<<<dim3(D_HEAD / 64, SEQ / 16, BATCH), 32, 0, stream>>>(xh, wkt, bK, khp, D_HEAD, 0);
  k_proj<<<dim3(D_MODEL / 64, SEQ / 16, BATCH), 32, 0, stream>>>(xh, wovt, bOV, vtp, D_MODEL, 1);

  // Pass 2: fused causal flash attention (WMMA + TDM V staging)
  const size_t smem_bytes = (size_t)2 * D_MODEL * KBLK * sizeof(bf16_t);  // 128 KB
  k_attn<<<dim3(SEQ / QTILE, BATCH), 256, smem_bytes, stream>>>(qhp, khp, vtp, out);
}
